// AttentionHead_88656714924527
// MI455X (gfx1250) — matrix operand order repaired and checked by reading, not compiled, not run on hardware
//
#include <hip/hip_runtime.h>

// ---------------------------------------------------------------------------
// Types for CDNA5 WMMA (gfx1250, wave32)
// ---------------------------------------------------------------------------
typedef __attribute__((ext_vector_type(16))) __bf16 v16bf;
typedef __attribute__((ext_vector_type(8)))  __bf16 v8bf;
typedef __attribute__((ext_vector_type(8)))  float  v8f;

union Frag { v16bf v; v8bf h[2]; };

static __device__ inline v8f zero8() {
    v8f z = {0.f,0.f,0.f,0.f,0.f,0.f,0.f,0.f};
    return z;
}

static __device__ inline v8f wmma_bf16(v16bf a, v16bf b, v8f c) {
    // D = A(16x32 bf16) x B(32x16 bf16) + C(16x16 f32)
    return __builtin_amdgcn_wmma_f32_16x16x32_bf16(
        /*neg_a=*/false, a, /*neg_b=*/false, b,
        /*c_mod=*/(short)0, c, /*reuse_a=*/false, /*reuse_b=*/false);
}

// Load one k-step's fragments: A (16x32) + 4x B (32x16), all bf16.
static __device__ inline void ld_frags(const __bf16* __restrict__ Ap,
                                       const __bf16* __restrict__ Bt,
                                       size_t ldb, int n0, int mr, int hf,
                                       int k, Frag& a, Frag b[4]) {
    a.h[0] = *(const v8bf*)(Ap + k + 8 * hf);
    a.h[1] = *(const v8bf*)(Ap + k + 16 + 8 * hf);
#pragma unroll
    for (int t = 0; t < 4; t++) {
        const __bf16* Bp = Bt + (size_t)(n0 + t * 16 + mr) * ldb + k + 8 * hf;
        b[t].h[0] = *(const v8bf*)Bp;
        b[t].h[1] = *(const v8bf*)(Bp + 16);
    }
}

// Core: one wave computes a 16 x 64 tile (4 accumulators), software-pipelined:
// next k-step's fragments are loaded while current k-step's 4 WMMAs execute.
static __device__ inline void gemm_core(const __bf16* __restrict__ Ap,
                                        const __bf16* __restrict__ Bt,
                                        size_t ldb, int n0, int mr, int hf,
                                        int k0, int k1, v8f acc[4]) {
    Frag a0, b0[4];
    ld_frags(Ap, Bt, ldb, n0, mr, hf, k0, a0, b0);
#pragma unroll 2
    for (int k = k0; k < k1; k += 32) {
        Frag a1, b1[4];
        int kn = k + 32;
        if (kn < k1) {
            __builtin_prefetch(Ap + kn + 32, 0, 1);   // global_prefetch_b8 on A
            ld_frags(Ap, Bt, ldb, n0, mr, hf, kn, a1, b1);
        }
#pragma unroll
        for (int t = 0; t < 4; t++) acc[t] = wmma_bf16(a0.v, b0[t].v, acc[t]);
        a0 = a1;
        b0[0] = b1[0]; b0[1] = b1[1]; b0[2] = b1[2]; b0[3] = b1[3];
    }
}

// ---------------------------------------------------------------------------
// Problem constants
// ---------------------------------------------------------------------------
#define BB     128
#define C_IN   1024
#define HWD    14
#define NTOK   196
#define DM     384
#define NH     12
#define HD     32
#define DFF    1536
#define MROWS  (BB * NTOK)          // 25088
#define BH     (BB * NH)            // 1536
#define KFIN   (NTOK * DM)          // 75264
#define JPAD   224                  // padded key length (7 * 32)
#define KSLICES 16
#define KSLICE_LEN (KFIN / KSLICES) // 4704

// ---------------------------------------------------------------------------
// Elementwise / prep kernels
// ---------------------------------------------------------------------------
__global__ __launch_bounds__(256)
void k_w2bf(const float* __restrict__ src, __bf16* __restrict__ dst, long n) {
    long idx = (long)blockIdx.x * 256 + threadIdx.x;
    if (idx < n) dst[idx] = (__bf16)src[idx];
}

__global__ __launch_bounds__(256)
void k_xt(const float* __restrict__ x, __bf16* __restrict__ xt) {
    long idx = (long)blockIdx.x * 256 + threadIdx.x;
    if (idx >= (long)MROWS * C_IN) return;
    int  c  = (int)(idx & (C_IN - 1));
    long bn = idx >> 10;            // b*196 + n
    int  n  = (int)(bn % NTOK);
    long b  = bn / NTOK;
    xt[idx] = (__bf16)x[(b * C_IN + c) * NTOK + n];
}

__global__ __launch_bounds__(256)
void k_conv(const float* __restrict__ t0, const float* __restrict__ w,
            const float* __restrict__ bias, float* __restrict__ t) {
    long idx = (long)blockIdx.x * 256 + threadIdx.x;
    if (idx >= (long)MROWS * DM) return;
    int  d  = (int)(idx % DM);
    long bn = idx / DM;
    int  n  = (int)(bn % NTOK);
    long b  = bn / NTOK;
    int  h  = n / HWD, wc = n % HWD;
    float acc = bias[d] + t0[idx];
#pragma unroll
    for (int kh = 0; kh < 3; kh++) {
        int hh = h + kh - 1;
        if (hh < 0 || hh > HWD - 1) continue;
#pragma unroll
        for (int kw = 0; kw < 3; kw++) {
            int ww = wc + kw - 1;
            if (ww < 0 || ww > HWD - 1) continue;
            acc += w[d * 9 + kh * 3 + kw] *
                   t0[(b * NTOK + hh * HWD + ww) * DM + d];
        }
    }
    t[idx] = acc;
}

__global__ __launch_bounds__(256)
void k_ln(const float* __restrict__ x, const float* __restrict__ w,
          const float* __restrict__ b, __bf16* __restrict__ out, int rows) {
    int row = blockIdx.x * 8 + (threadIdx.x >> 5);
    if (row >= rows) return;
    int lane = threadIdx.x & 31;
    const float* xr = x + (size_t)row * DM;
    float s = 0.f, s2 = 0.f;
    for (int i = lane; i < DM; i += 32) { float v = xr[i]; s += v; s2 += v * v; }
    for (int o = 16; o > 0; o >>= 1) {
        s  += __shfl_xor(s,  o, 32);
        s2 += __shfl_xor(s2, o, 32);
    }
    float mu  = s  * (1.f / DM);
    float var = s2 * (1.f / DM) - mu * mu;
    float rs  = rsqrtf(var + 1e-5f);
    __bf16* orow = out + (size_t)row * DM;
    for (int i = lane; i < DM; i += 32)
        orow[i] = (__bf16)((xr[i] - mu) * rs * w[i] + b[i]);
}

// piecewise relative-position bucket (ALPHA=1.9, BETA=3.8, GAMMA=15.2)
static __device__ inline int pw_idx(int d) {
    float ad = fabsf((float)d);
    if (ad <= 1.9f) return d;
    float y = 1.9f + logf(ad / 1.9f) / logf(15.2f / 1.9f) * (3.8f - 1.9f);
    float r = fminf(rintf(y), 3.8f);
    return (d > 0 ? 1 : -1) * (int)r;
}

__global__ __launch_bounds__(256)
void k_rp(int* __restrict__ rp) {
    int idx = blockIdx.x * 256 + threadIdx.x;
    if (idx >= NTOK * NTOK) return;
    int i = idx / NTOK, j = idx % NTOK;
    int hi = i / HWD, wi = i % HWD, hj = j / HWD, wj = j % HWD;
    rp[idx] = (pw_idx(hi - hj) + 3) * 7 + (pw_idx(wi - wj) + 3);
}

// ---------------------------------------------------------------------------
// Generic WMMA GEMM kernels (B pre-converted to bf16)
// ---------------------------------------------------------------------------
// EPI: 0 = +bias, relu6 clip, bf16 out;  1 = +bias, f32 out;
//      2 = +bias, +residual, f32 out
template <int EPI>
__global__ __launch_bounds__(256)
void k_gemm(const __bf16* __restrict__ A, const __bf16* __restrict__ Bt,
            const float* __restrict__ bias, const float* __restrict__ res,
            float* __restrict__ outF, __bf16* __restrict__ outB,
            int M, int Nn, int K, int lda, int ldb) {
    int Ng   = Nn >> 6;
    int wave = blockIdx.x * 8 + (threadIdx.x >> 5);
    if (wave >= (M >> 4) * Ng) return;
    int mt = wave / Ng, ng = wave - mt * Ng;
    int m0 = mt << 4, n0 = ng << 6;
    int lane = threadIdx.x & 31, mr = lane & 15, hf = lane >> 4;
    const __bf16* Ap = A + (size_t)(m0 + mr) * lda;
    v8f acc[4] = { zero8(), zero8(), zero8(), zero8() };
    gemm_core(Ap, Bt, (size_t)ldb, n0, mr, hf, 0, K, acc);
#pragma unroll
    for (int t = 0; t < 4; t++) {
        int n = n0 + t * 16 + mr;
        float bv = bias[n];
#pragma unroll
        for (int r = 0; r < 8; r++) {
            int m = m0 + r + 8 * hf;
            float v = acc[t][r] + bv;
            if (EPI == 2) v += res[(size_t)m * Nn + n];
            if (EPI == 0) {
                v = fminf(fmaxf(v, 0.f), 6.f);
                outB[(size_t)m * Nn + n] = (__bf16)v;
            } else {
                outF[(size_t)m * Nn + n] = v;
            }
        }
    }
}

// qkv GEMM: scatters scaled q, k, and transposed+padded v^T
__global__ __launch_bounds__(256)
void k_gemm_qkv(const __bf16* __restrict__ A, const __bf16* __restrict__ Bt,
                __bf16* __restrict__ q, __bf16* __restrict__ kk,
                __bf16* __restrict__ vt) {
    const int Ng = 18;  // 1152 / 64
    int wave = blockIdx.x * 8 + (threadIdx.x >> 5);
    if (wave >= (MROWS >> 4) * Ng) return;
    int mt = wave / Ng, ng = wave - mt * Ng;
    int m0 = mt << 4, n0 = ng << 6;
    int lane = threadIdx.x & 31, mr = lane & 15, hf = lane >> 4;
    const __bf16* Ap = A + (size_t)(m0 + mr) * DM;
    v8f acc[4] = { zero8(), zero8(), zero8(), zero8() };
    gemm_core(Ap, Bt, DM, n0, mr, hf, 0, DM, acc);
#pragma unroll
    for (int t = 0; t < 4; t++) {
        int n = n0 + t * 16 + mr;
        int which = n / DM;
        int hd = n - which * DM;
        int head = hd >> 5, d = hd & 31;
#pragma unroll
        for (int r = 0; r < 8; r++) {
            int m = m0 + r + 8 * hf;
            int b_ = m / NTOK, i = m - b_ * NTOK;
            size_t bh = (size_t)(b_ * NH + head);
            float v = acc[t][r];
            if (which == 0)
                q[(bh * NTOK + i) * HD + d] = (__bf16)(v * 0.17677669529663687f);
            else if (which == 1)
                kk[(bh * NTOK + i) * HD + d] = (__bf16)v;
            else
                vt[bh * HD * JPAD + (size_t)d * JPAD + i] = (__bf16)v;
        }
    }
}

// split-K final GEMM: M=128, N=384, K=75264 -> 16 deterministic partials
__global__ __launch_bounds__(256)
void k_gemm_splitk(const __bf16* __restrict__ A, const __bf16* __restrict__ Bt,
                   float* __restrict__ Yp) {
    int wave = blockIdx.x * 8 + (threadIdx.x >> 5);
    if (wave >= KSLICES * 8 * 6) return;
    int ks  = wave / 48;
    int rem = wave - ks * 48;
    int mt = rem / 6, ng = rem - mt * 6;
    int m0 = mt << 4, n0 = ng << 6;
    int lane = threadIdx.x & 31, mr = lane & 15, hf = lane >> 4;
    const __bf16* Ap = A + (size_t)(m0 + mr) * KFIN;
    v8f acc[4] = { zero8(), zero8(), zero8(), zero8() };
    int k0 = ks * KSLICE_LEN;
    gemm_core(Ap, Bt, KFIN, n0, mr, hf, k0, k0 + KSLICE_LEN, acc);
    float* Yk = Yp + (size_t)ks * (BB * DM);
#pragma unroll
    for (int t = 0; t < 4; t++) {
        int n = n0 + t * 16 + mr;
#pragma unroll
        for (int r = 0; r < 8; r++) {
            int m = m0 + r + 8 * hf;
            Yk[(size_t)m * DM + n] = acc[t][r];
        }
    }
}

// ---------------------------------------------------------------------------
// Fused attention: S = q k^T (WMMA), + RPE bucket bias, softmax, O = P v (WMMA)
// grid = (13 m-tiles, 1536 batch-heads), block = 128 (4 waves)
// ---------------------------------------------------------------------------
__global__ __launch_bounds__(128)
void k_attn(const __bf16* __restrict__ Q, const __bf16* __restrict__ Kb,
            const __bf16* __restrict__ VT, const float* __restrict__ rpe,
            const int* __restrict__ rp, __bf16* __restrict__ O) {
    __shared__ float  S[16][208];
    __shared__ float  QL[16][32];
    __shared__ float  RPE[32][52];
    __shared__ float  BS[16][56];
    __shared__ float  RED[16][8];
    __shared__ __bf16 P[16][JPAD];

    const int bh = blockIdx.y;
    const int m0 = blockIdx.x * 16;
    const int tid = threadIdx.x;
    const int wv = tid >> 5, lane = tid & 31, mr = lane & 15, hf = lane >> 4;

    const __bf16* qb = Q  + (size_t)bh * NTOK * HD;
    const __bf16* kb = Kb + (size_t)bh * NTOK * HD;

    for (int i = tid; i < 32 * 49; i += 128) RPE[i / 49][i % 49] = rpe[i];
    for (int i = tid; i < 16 * 32; i += 128) {
        int r = i >> 5, c = i & 31, row = m0 + r;
        QL[r][c] = (row < NTOK) ? (float)qb[(size_t)row * HD + c] : 0.f;
    }
    __syncthreads();

    // ---- phase 1: S tile rows (16 x 208) via WMMA, K = 32 ----
    for (int nt = wv; nt < 13; nt += 4) {
        int n0 = nt * 16;
        Frag a, b;
        const __bf16* ap = qb + (size_t)(m0 + mr) * HD;
        a.h[0] = *(const v8bf*)(ap + 8 * hf);
        a.h[1] = *(const v8bf*)(ap + 16 + 8 * hf);
        const __bf16* kp = kb + (size_t)(n0 + mr) * HD;
        b.h[0] = *(const v8bf*)(kp + 8 * hf);
        b.h[1] = *(const v8bf*)(kp + 16 + 8 * hf);
        v8f acc = zero8();
        acc = wmma_bf16(a.v, b.v, acc);
#pragma unroll
        for (int r = 0; r < 8; r++) S[r + 8 * hf][n0 + mr] = acc[r];
    }
    __syncthreads();

    // ---- phase 2: bias + softmax (8 threads per row) ----
    const int r = tid >> 3, sub = tid & 7;
    const int irow = m0 + r;
    for (int u = sub; u < 49; u += 8) {
        float s = 0.f;
#pragma unroll
        for (int d = 0; d < 32; d++) s += QL[r][d] * RPE[d][u];
        BS[r][u] = s;
    }
    __syncthreads();
    float mx = -1e30f;
    if (irow < NTOK) {
        const int* rpr = rp + irow * NTOK;
        for (int j = sub; j < NTOK; j += 8) {
            float v = S[r][j] + BS[r][rpr[j]];
            S[r][j] = v;
            mx = fmaxf(mx, v);
        }
    }
    RED[r][sub] = mx;
    __syncthreads();
    mx = RED[r][0];
#pragma unroll
    for (int t = 1; t < 8; t++) mx = fmaxf(mx, RED[r][t]);
    float sm = 0.f;
    if (irow < NTOK) {
        for (int j = sub; j < NTOK; j += 8) {
            float e = __expf(S[r][j] - mx);
            P[r][j] = (__bf16)e;
            sm += e;
        }
    } else {
        for (int j = sub; j < NTOK; j += 8) P[r][j] = (__bf16)0.f;
    }
    for (int j = NTOK + sub; j < JPAD; j += 8) P[r][j] = (__bf16)0.f;
    __syncthreads();
    RED[r][sub] = sm;
    __syncthreads();
    sm = 0.f;
#pragma unroll
    for (int t = 0; t < 8; t++) sm += RED[r][t];
    if (irow < NTOK) {
        float rinv = 1.f / sm;
        for (int j = sub; j < NTOK; j += 8)
            P[r][j] = (__bf16)((float)P[r][j] * rinv);
    }
    __syncthreads();

    // ---- phase 3: O = P (16x224) @ v^T, waves 0/1 cover d 0-15 / 16-31 ----
    if (wv < 2) {
        int n0 = wv * 16;
        const __bf16* vb = VT + (size_t)bh * HD * JPAD;
        v8f acc = zero8();
#pragma unroll
        for (int k = 0; k < JPAD; k += 32) {
            Frag a, b;
            a.h[0] = *(const v8bf*)(&P[mr][k + 8 * hf]);
            a.h[1] = *(const v8bf*)(&P[mr][k + 16 + 8 * hf]);
            const __bf16* vp = vb + (size_t)(n0 + mr) * JPAD + k;
            b.h[0] = *(const v8bf*)(vp + 8 * hf);
            b.h[1] = *(const v8bf*)(vp + 16 + 8 * hf);
            acc = wmma_bf16(a.v, b.v, acc);
        }
        int b_ = bh / NH, h_ = bh - b_ * NH;
        int dcol = n0 + mr;
#pragma unroll
        for (int rr = 0; rr < 8; rr++) {
            int i = m0 + rr + 8 * hf;
            if (i < NTOK)
                O[((size_t)(b_ * NTOK + i)) * DM + h_ * HD + dcol] = (__bf16)acc[rr];
        }
    }
}

// ---------------------------------------------------------------------------
// BatchNorm kernels
// ---------------------------------------------------------------------------
__global__ __launch_bounds__(256)
void k_bn1(const float* __restrict__ t3, const float* __restrict__ w,
           const float* __restrict__ bb, float* __restrict__ sn,
           float* __restrict__ cn) {
    int n = blockIdx.x;                 // 0..195
    __shared__ float rs[256], rs2[256];
    float s = 0.f, s2 = 0.f;
    for (int i = threadIdx.x; i < BB * DM; i += 256) {
        int b = i / DM, d = i - (i / DM) * DM;
        float v = t3[((size_t)(b * NTOK + n)) * DM + d];
        s += v; s2 += v * v;
    }
    rs[threadIdx.x] = s; rs2[threadIdx.x] = s2;
    __syncthreads();
    for (int o = 128; o > 0; o >>= 1) {
        if (threadIdx.x < o) {
            rs[threadIdx.x]  += rs[threadIdx.x + o];
            rs2[threadIdx.x] += rs2[threadIdx.x + o];
        }
        __syncthreads();
    }
    if (threadIdx.x == 0) {
        const float inv = 1.f / (BB * DM);
        float mu  = rs[0] * inv;
        float var = rs2[0] * inv - mu * mu;
        float sc  = w[n] * rsqrtf(var + 2e-5f);
        sn[n] = sc;
        cn[n] = bb[n] - mu * sc;
    }
}

__global__ __launch_bounds__(256)
void k_t4(const float* __restrict__ t3, const float* __restrict__ sn,
          const float* __restrict__ cn, __bf16* __restrict__ t4) {
    long idx = (long)blockIdx.x * 256 + threadIdx.x;
    if (idx >= (long)BB * KFIN) return;
    int n = (int)((idx / DM) % NTOK);
    t4[idx] = (__bf16)(t3[idx] * sn[n] + cn[n]);
}

__global__ __launch_bounds__(128)
void k_bn2(const float* __restrict__ Yp, const float* __restrict__ w,
           const float* __restrict__ bb, float* __restrict__ out) {
    int j = blockIdx.x;                 // column 0..383
    int t = threadIdx.x;                // row 0..127
    float v = 0.f;
#pragma unroll
    for (int ks = 0; ks < KSLICES; ks++)
        v += Yp[(size_t)ks * (BB * DM) + (size_t)t * DM + j];
    __shared__ float rs[128], rs2[128];
    rs[t] = v; rs2[t] = v * v;
    __syncthreads();
    for (int o = 64; o > 0; o >>= 1) {
        if (t < o) { rs[t] += rs[t + o]; rs2[t] += rs2[t + o]; }
        __syncthreads();
    }
    float mu  = rs[0] * (1.f / BB);
    float var = rs2[0] * (1.f / BB) - mu * mu;
    out[(size_t)t * DM + j] = (v - mu) * rsqrtf(var + 2e-5f) * w[j] + bb[j];
}

// ---------------------------------------------------------------------------
// Host launcher
// ---------------------------------------------------------------------------
extern "C" void kernel_launch(void* const* d_in, const int* in_sizes, int n_in,
                              void* d_out, int out_size, void* d_ws,
                              size_t ws_size, hipStream_t stream) {
    (void)in_sizes; (void)n_in; (void)out_size;
    const float* x        = (const float*)d_in[0];
    const float* fc1_w    = (const float*)d_in[1];
    const float* fc1_b    = (const float*)d_in[2];
    const float* fc2_w    = (const float*)d_in[3];
    const float* fc2_b    = (const float*)d_in[4];
    const float* conv_w   = (const float*)d_in[5];
    const float* conv_b   = (const float*)d_in[6];
    const float* ln1_w    = (const float*)d_in[7];
    const float* ln1_b    = (const float*)d_in[8];
    const float* qkv_w    = (const float*)d_in[9];
    const float* rpe_tab  = (const float*)d_in[10];
    const float* proj_w   = (const float*)d_in[11];
    const float* proj_b   = (const float*)d_in[12];
    const float* ln2_w    = (const float*)d_in[13];
    const float* ln2_b    = (const float*)d_in[14];
    const float* mfc1_w   = (const float*)d_in[15];
    const float* mfc1_b   = (const float*)d_in[16];
    const float* mfc2_w   = (const float*)d_in[17];
    const float* mfc2_b   = (const float*)d_in[18];
    const float* bn1_w    = (const float*)d_in[19];
    const float* bn1_b    = (const float*)d_in[20];
    const float* lin_w    = (const float*)d_in[21];
    /* lin_b (d_in[22]) cancels under BN2 */
    const float* bn2_w    = (const float*)d_in[23];
    const float* bn2_b    = (const float*)d_in[24];
    float* out = (float*)d_out;

    char* ws = (char*)d_ws;
    size_t off = 0;
    auto alloc = [&](size_t bytes) {
        size_t o = off;
        off += (bytes + 255) & ~(size_t)255;
        return o;
    };

    const size_t XT_BYTES  = (size_t)MROWS * C_IN * 2;            // 51.4 MB
    const size_t BIG_BYTES = 2 * XT_BYTES;                        // also holds H1 (77 MB)
    size_t o_big  = alloc(BIG_BYTES);
    size_t o_t0   = alloc((size_t)MROWS * DM * 4);
    size_t o_t    = alloc((size_t)MROWS * DM * 4);
    size_t o_h    = alloc((size_t)MROWS * DM * 2);                // also t4 (same size)
    size_t o_q    = alloc(((size_t)BH * NTOK + 16) * HD * 2);
    size_t o_k    = alloc(((size_t)BH * NTOK + 16) * HD * 2);
    size_t o_vt   = alloc((size_t)BH * HD * JPAD * 2);
    size_t o_o    = alloc((size_t)MROWS * DM * 2);
    size_t o_rp   = alloc((size_t)NTOK * NTOK * 4);
    size_t o_sn   = alloc(NTOK * 4);
    size_t o_cn   = alloc(NTOK * 4);
    size_t o_yp   = alloc((size_t)KSLICES * BB * DM * 4);
    // bf16 weight copies
    size_t o_w1   = alloc((size_t)1024 * C_IN * 2);
    size_t o_w2   = alloc((size_t)DM * C_IN * 2);
    size_t o_wqkv = alloc((size_t)3 * DM * DM * 2);
    size_t o_wpr  = alloc((size_t)DM * DM * 2);
    size_t o_wm1  = alloc((size_t)DFF * DM * 2);
    size_t o_wm2  = alloc((size_t)DM * DFF * 2);
    size_t o_wlin = alloc((size_t)DM * KFIN * 2);
    if (off > ws_size) return;

    __bf16* XT   = (__bf16*)(ws + o_big);
    __bf16* A1   = (__bf16*)(ws + o_big + XT_BYTES);
    __bf16* H1   = (__bf16*)(ws + o_big);                         // reuse for MLP hidden
    float*  T0   = (float*)(ws + o_t0);
    float*  T    = (float*)(ws + o_t);
    __bf16* H    = (__bf16*)(ws + o_h);
    __bf16* T4   = (__bf16*)(ws + o_h);                           // reuse after H is done
    __bf16* Qb   = (__bf16*)(ws + o_q);
    __bf16* Kbf  = (__bf16*)(ws + o_k);
    __bf16* VT   = (__bf16*)(ws + o_vt);
    __bf16* Ob   = (__bf16*)(ws + o_o);
    int*    RP   = (int*)(ws + o_rp);
    float*  SN   = (float*)(ws + o_sn);
    float*  CN   = (float*)(ws + o_cn);
    float*  YP   = (float*)(ws + o_yp);
    __bf16* W1   = (__bf16*)(ws + o_w1);
    __bf16* W2   = (__bf16*)(ws + o_w2);
    __bf16* WQKV = (__bf16*)(ws + o_wqkv);
    __bf16* WPR  = (__bf16*)(ws + o_wpr);
    __bf16* WM1  = (__bf16*)(ws + o_wm1);
    __bf16* WM2  = (__bf16*)(ws + o_wm2);
    __bf16* WLIN = (__bf16*)(ws + o_wlin);

    hipMemsetAsync(VT, 0, (size_t)BH * HD * JPAD * 2, stream);

    // 0. convert weights to bf16 once per call
    auto conv_w2bf = [&](const float* src, __bf16* dst, long n) {
        k_w2bf<<<(unsigned)((n + 255) / 256), 256, 0, stream>>>(src, dst, n);
    };
    conv_w2bf(fc1_w,  W1,   (long)1024 * C_IN);
    conv_w2bf(fc2_w,  W2,   (long)DM * C_IN);
    conv_w2bf(qkv_w,  WQKV, (long)3 * DM * DM);
    conv_w2bf(proj_w, WPR,  (long)DM * DM);
    conv_w2bf(mfc1_w, WM1,  (long)DFF * DM);
    conv_w2bf(mfc2_w, WM2,  (long)DM * DFF);
    conv_w2bf(lin_w,  WLIN, (long)DM * KFIN);

    // 1. x -> token-major bf16
    {
        long total = (long)MROWS * C_IN;
        k_xt<<<(unsigned)((total + 255) / 256), 256, 0, stream>>>(x, XT);
    }
    // 2. embed fc1 (K=1024, N=1024) + relu6 -> bf16
    k_gemm<0><<<(MROWS / 16) * (1024 / 64) / 8, 256, 0, stream>>>(
        XT, W1, fc1_b, nullptr, nullptr, A1, MROWS, 1024, 1024, 1024, 1024);
    // 3. embed fc2 (K=1024, N=384) -> f32 t0
    k_gemm<1><<<(MROWS / 16) * (DM / 64) / 8, 256, 0, stream>>>(
        A1, W2, fc2_b, nullptr, T0, nullptr, MROWS, DM, 1024, 1024, 1024);
    // 4. depthwise conv + bias + residual -> t
    {
        long total = (long)MROWS * DM;
        k_conv<<<(unsigned)((total + 255) / 256), 256, 0, stream>>>(T0, conv_w, conv_b, T);
    }
    // 5. LN1 -> bf16 h
    k_ln<<<MROWS / 8, 256, 0, stream>>>(T, ln1_w, ln1_b, H, MROWS);
    // 6. qkv (K=384, N=1152) -> q (scaled), k, v^T (padded)
    k_gemm_qkv<<<(MROWS / 16) * 18 / 8, 256, 0, stream>>>(H, WQKV, Qb, Kbf, VT);
    // 7. relative-position buckets
    k_rp<<<(NTOK * NTOK + 255) / 256, 256, 0, stream>>>(RP);
    // 8. fused attention -> O bf16 (B, N, 384)
    {
        dim3 g(13, BH);
        k_attn<<<g, 128, 0, stream>>>(Qb, Kbf, VT, rpe_tab, RP, Ob);
    }
    // 9. proj + bias + residual(t) -> t2 (reuses T0)
    k_gemm<2><<<(MROWS / 16) * (DM / 64) / 8, 256, 0, stream>>>(
        Ob, WPR, proj_b, T, T0, nullptr, MROWS, DM, DM, DM, DM);
    // 10. LN2 -> bf16 h
    k_ln<<<MROWS / 8, 256, 0, stream>>>(T0, ln2_w, ln2_b, H, MROWS);
    // 11. MLP fc1 (K=384, N=1536) + relu6 -> bf16 h1
    k_gemm<0><<<(MROWS / 16) * (DFF / 64) / 8, 256, 0, stream>>>(
        H, WM1, mfc1_b, nullptr, nullptr, H1, MROWS, DFF, DM, DM, DM);
    // 12. MLP fc2 (K=1536, N=384) + bias + residual(t2) -> t3 (reuses T)
    k_gemm<2><<<(MROWS / 16) * (DM / 64) / 8, 256, 0, stream>>>(
        H1, WM2, mfc2_b, T0, T, nullptr, MROWS, DM, DFF, DFF, DFF);
    // 13. BN1 stats per position n
    k_bn1<<<NTOK, 256, 0, stream>>>(T, bn1_w, bn1_b, SN, CN);
    // 14. normalized t4 -> bf16 (128 x 75264)
    {
        long total = (long)BB * KFIN;
        k_t4<<<(unsigned)((total + 255) / 256), 256, 0, stream>>>(T, SN, CN, T4);
    }
    // 15. final linear, split-K (16 partials)
    k_gemm_splitk<<<(KSLICES * 8 * 6) / 8, 256, 0, stream>>>(T4, WLIN, YP);
    // 16. reduce partials + BN2 -> output
    k_bn2<<<DM, 128, 0, stream>>>(YP, bn2_w, bn2_b, out);
}
